// HeatTransferNetworkInterpolate_28767690948633
// MI455X (gfx1250) — matrix operand order repaired and checked by reading, not compiled, not run on hardware
//
#include <hip/hip_runtime.h>

// ---------------------------------------------------------------------------
// Types for CDNA5 WMMA (wave32): v_wmma_f32_16x16x32_bf16
// ---------------------------------------------------------------------------
typedef __attribute__((ext_vector_type(16))) __bf16 v16bf;
typedef __attribute__((ext_vector_type(8)))  __bf16 v8bf;
typedef __attribute__((ext_vector_type(8)))  float  v8f;
typedef __attribute__((ext_vector_type(4)))  int    v4i;

struct bf16x16 { v8bf lo, hi; };

static __device__ inline __bf16 f2bf(float f) { return (__bf16)f; }

static __device__ inline v8bf pack8(float4 a, float4 b) {
    v8bf v;
    v[0] = f2bf(a.x); v[1] = f2bf(a.y); v[2] = f2bf(a.z); v[3] = f2bf(a.w);
    v[4] = f2bf(b.x); v[5] = f2bf(b.y); v[6] = f2bf(b.z); v[7] = f2bf(b.w);
    return v;
}

// 16 contiguous bf16 (32B, 16B-aligned) -> one WMMA operand (2x ds_load_b128)
static __device__ inline v16bf ld_frag16(const __bf16* p) {
    bf16x16 t;
    t.lo = *(const v8bf*)p;
    t.hi = *(const v8bf*)(p + 8);
    return __builtin_bit_cast(v16bf, t);
}

// A fragment: two contiguous 8-element runs at p and p+16 (both 16B-aligned)
static __device__ inline v16bf ld_frag8x2(const __bf16* p) {
    bf16x16 t;
    t.lo = *(const v8bf*)p;
    t.hi = *(const v8bf*)(p + 16);
    return __builtin_bit_cast(v16bf, t);
}

// Async 16B global->LDS copy (gfx1250 GLOBAL_LOAD_ASYNC_TO_LDS_B128, ASYNCcnt)
#if __has_builtin(__builtin_amdgcn_global_load_async_to_lds_b128)
#define HAVE_ASYNC_LDS 1
static __device__ inline void async_cp16(void* g, void* l) {
    __builtin_amdgcn_global_load_async_to_lds_b128(
        (__attribute__((address_space(1))) v4i*)g,
        (__attribute__((address_space(3))) v4i*)l, 0, 0);
}
#else
#define HAVE_ASYNC_LDS 0
#endif

// ---------------------------------------------------------------------------
// Stage 0: one-time weight prep -> bf16, WMMA B-fragment order
//   frag layout: [kk][nt][lane][16j];  b[j] = W[(kk*32 + j + 16*lh)*64 + nt*16 + M]
//   wf[0:8192) = W1 (K=128), wf[8192:12288) = W2 (K=64)
// ---------------------------------------------------------------------------
__global__ __launch_bounds__(256) void k_wprep(const float* __restrict__ w1,
                                               const float* __restrict__ w2,
                                               __bf16* __restrict__ wf) {
    int tid = threadIdx.x;
    for (int i = tid; i < 128 * 64; i += 256) {
        int K = i >> 6, col = i & 63;
        int kk = K >> 5, rb = K & 31;
        int lh = rb >> 4, j = rb & 15;
        int nt = col >> 4, M = col & 15;
        int lane = lh * 16 + M;
        wf[(((kk * 4 + nt) * 32 + lane) << 4) | j] = f2bf(w1[i]);
    }
    for (int i = tid; i < 64 * 64; i += 256) {
        int K = i >> 6, col = i & 63;
        int kk = K >> 5, rb = K & 31;
        int lh = rb >> 4, j = rb & 15;
        int nt = col >> 4, M = col & 15;
        int lane = lh * 16 + M;
        wf[8192 + ((((kk * 4 + nt) * 32 + lane) << 4) | j)] = f2bf(w2[i]);
    }
}

// ---------------------------------------------------------------------------
// Stage 1: xs = relu([x, pos] @ lin_sim_w + b)   (N x 3) @ (3 x 64)
// ---------------------------------------------------------------------------
__global__ void k_xs(const float* __restrict__ x, const float* __restrict__ pos,
                     const float* __restrict__ w, const float* __restrict__ b,
                     int N, float* __restrict__ xs) {
    int i = blockIdx.x * blockDim.x + threadIdx.x;
    if (i >= N * 64) return;
    int n = i >> 6, h = i & 63;
    float v = x[n] * w[h] + pos[n * 2] * w[64 + h] + pos[n * 2 + 1] * w[128 + h] + b[h];
    xs[i] = v > 0.f ? v : 0.f;
}

// ---------------------------------------------------------------------------
// Stage 2: EdgeConv via WMMA.  Per wave: 16 edges.
//   z[e] = [xs[dst], xs[src]-xs[dst]]  (128)
//   m = relu(z @ W1 + b1) @ W2 + b2    -> atomic max into agg[dst] (agg init 0)
// Pre-swizzled bf16 weights stream into LDS with async global->LDS b128 copies.
// ---------------------------------------------------------------------------
__global__ __launch_bounds__(128) void k_edgeconv(
    const float* __restrict__ xs, const int* __restrict__ esrc,
    const int* __restrict__ edst, int E,
    const __bf16* __restrict__ wf,
    const float* __restrict__ b1, const float* __restrict__ b2,
    float* __restrict__ agg)
{
    __shared__ __align__(16) __bf16 sWf[12288];         // W1f (8192) + W2f (4096)
    __shared__ __align__(16) __bf16 sZ[4][16 * 128];    // per-wave z, row-major
    __shared__ __align__(16) __bf16 sM[4][16 * 64];     // per-wave m, row-major

    const int tid = threadIdx.x;
    // 24KB contiguous weight image -> LDS
#if HAVE_ASYNC_LDS
    {
        char* g = (char*)wf;
        char* l = (char*)sWf;
        for (int o = tid * 16; o < 24576; o += 128 * 16)
            async_cp16(g + o, l + o);
        asm volatile("s_wait_asynccnt 0x0" ::: "memory");
    }
#else
    {
        const uint4* g = (const uint4*)wf;
        uint4* l = (uint4*)sWf;
        for (int o = tid; o < 24576 / 16; o += 128) l[o] = g[o];
    }
#endif

    const int wave = tid >> 5, lane = tid & 31;
    const int le = lane & 15, lh = lane >> 4;   // edge-in-tile, lane half
    const int ebase = blockIdx.x * 64 + wave * 16;
    const int e = ebase + le;

    // stage z as bf16, 8 x ds_store_b128 per lane
    v8bf* dstv = (v8bf*)&sZ[wave][le * 128 + lh * 64];
    if (e < E) {
        const float4* xd4 = (const float4*)(xs + (size_t)edst[e] * 64);
        if (lh == 0) {
#pragma unroll
            for (int c = 0; c < 8; ++c) dstv[c] = pack8(xd4[2 * c], xd4[2 * c + 1]);
        } else {
            const float4* xs4 = (const float4*)(xs + (size_t)esrc[e] * 64);
#pragma unroll
            for (int c = 0; c < 8; ++c) {
                float4 a = xs4[2 * c], b = xs4[2 * c + 1];
                float4 da = xd4[2 * c], db = xd4[2 * c + 1];
                float4 r0 = make_float4(a.x - da.x, a.y - da.y, a.z - da.z, a.w - da.w);
                float4 r1 = make_float4(b.x - db.x, b.y - db.y, b.z - db.z, b.w - db.w);
                dstv[c] = pack8(r0, r1);
            }
        }
    } else {
        float4 z4 = make_float4(0.f, 0.f, 0.f, 0.f);
#pragma unroll
        for (int c = 0; c < 8; ++c) dstv[c] = pack8(z4, z4);
    }
    __syncthreads();

    const int M = le;           // A row / result column (lane % 16)
    // ---- GEMM1: [16 x 128] @ [128 x 64] ----
    v8f acc[4] = {};
#pragma unroll
    for (int kk = 0; kk < 4; ++kk) {
        // A: 16 contiguous at K0, K0 = kk*32 + 16*lh
        v16bf a = ld_frag16(&sZ[wave][M * 128 + kk * 32 + 16 * lh]);
#pragma unroll
        for (int nt = 0; nt < 4; ++nt) {
            v16bf b = ld_frag16(&sWf[((kk * 4 + nt) * 32 + lane) << 4]);
            acc[nt] = __builtin_amdgcn_wmma_f32_16x16x32_bf16(
                false, a, false, b, (short)0, acc[nt], false, false);
        }
    }
    // bias + relu -> sM; C/D layout: elem r -> row = r + 8*lh, col = nt*16 + M
#pragma unroll
    for (int nt = 0; nt < 4; ++nt) {
        const int col = nt * 16 + M;
        const float bb = b1[col];
#pragma unroll
        for (int r = 0; r < 8; ++r) {
            const int row = r + 8 * lh;
            float v = acc[nt][r] + bb;
            sM[wave][row * 64 + col] = f2bf(v > 0.f ? v : 0.f);
        }
    }
    __syncthreads();

    // ---- GEMM2: [16 x 64] @ [64 x 64] ----
    v8f acc2[4] = {};
#pragma unroll
    for (int kk = 0; kk < 2; ++kk) {
        v16bf a = ld_frag16(&sM[wave][M * 64 + kk * 32 + 16 * lh]);
#pragma unroll
        for (int nt = 0; nt < 4; ++nt) {
            v16bf b = ld_frag16(&sWf[8192 + (((kk * 4 + nt) * 32 + lane) << 4)]);
            acc2[nt] = __builtin_amdgcn_wmma_f32_16x16x32_bf16(
                false, a, false, b, (short)0, acc2[nt], false, false);
        }
    }
    // bias + segment-max (agg init 0; positive float == int compare)
#pragma unroll
    for (int nt = 0; nt < 4; ++nt) {
        const int col = nt * 16 + M;
        const float bb = b2[col];
#pragma unroll
        for (int r = 0; r < 8; ++r) {
            const int row = r + 8 * lh;
            const int e2 = ebase + row;
            if (e2 < E) {
                float v = acc2[nt][r] + bb;
                if (v > 0.f)
                    atomicMax((int*)&agg[(size_t)edst[e2] * 64 + col], __float_as_int(v));
            }
        }
    }
}

// ---------------------------------------------------------------------------
// Stage 3: k-means (K=4, dim=64), tol-gated
// ---------------------------------------------------------------------------
__global__ void k_km_init(const float* __restrict__ xs2, const int* __restrict__ idx,
                          float* __restrict__ cent) {
    int i = threadIdx.x;                       // 256 == 4*64
    int k = i >> 6, h = i & 63;
    cent[i] = xs2[(size_t)idx[k] * 64 + h];
}

__global__ __launch_bounds__(256) void k_assign(
    const float* __restrict__ xs2, const float* __restrict__ cent, int N,
    int* __restrict__ assign, float* __restrict__ sums, float* __restrict__ cnts,
    const int* __restrict__ gate, int accumulate)
{
    if (gate && gate[0]) return;               // converged: cheap early exit
    __shared__ __align__(16) float sc[256];
    __shared__ float ssum[256];
    __shared__ float scnt[4];
    int tid = threadIdx.x;
    sc[tid] = cent[tid];
    if (accumulate) { ssum[tid] = 0.f; if (tid < 4) scnt[tid] = 0.f; }
    __syncthreads();
    int n = blockIdx.x * 256 + tid;
    if (n < N) {
        float4 xr[16];
        const float4* row = (const float4*)(xs2 + (size_t)n * 64);
#pragma unroll
        for (int h = 0; h < 16; ++h) xr[h] = row[h];
        float best = 3.4e38f; int bk = 0;
        for (int k = 0; k < 4; ++k) {
            const float4* c4 = (const float4*)&sc[k * 64];
            float d = 0.f;
#pragma unroll
            for (int h = 0; h < 16; ++h) {
                float4 c = c4[h];
                float t0 = xr[h].x - c.x, t1 = xr[h].y - c.y;
                float t2 = xr[h].z - c.z, t3 = xr[h].w - c.w;
                d += t0 * t0 + t1 * t1 + t2 * t2 + t3 * t3;
            }
            if (d < best) { best = d; bk = k; }
        }
        assign[n] = bk;
        if (accumulate) {
            const float* rowf = (const float*)xr;
            for (int h = 0; h < 64; ++h) atomicAdd(&ssum[bk * 64 + h], rowf[h]);
            atomicAdd(&scnt[bk], 1.f);
        }
    }
    __syncthreads();
    if (accumulate) {
        atomicAdd(&sums[tid], ssum[tid]);
        if (tid < 4) atomicAdd(&cnts[tid], scnt[tid]);
    }
}

__global__ __launch_bounds__(256) void k_update(
    float* __restrict__ cent, float* __restrict__ sums, float* __restrict__ cnts,
    int* __restrict__ gate, float tol)
{
    if (gate[0]) return;
    __shared__ float red[256];
    int tid = threadIdx.x;
    int k = tid >> 6;
    float c_old = cent[tid];
    float cnt = cnts[k];
    float nc = cnt > 0.f ? sums[tid] / cnt : c_old;   // empty cluster keeps centroid
    float d = nc - c_old;
    cent[tid] = nc;
    sums[tid] = 0.f;
    red[tid] = d * d;
    __syncthreads();
    for (int s = 128; s > 0; s >>= 1) {
        if (tid < s) red[tid] += red[tid + s];
        __syncthreads();
    }
    if (tid < 4) cnts[tid] = 0.f;
    if (tid == 0 && red[0] < tol * tol) gate[0] = 1;  // ||shift|| < tol
}

// ---------------------------------------------------------------------------
// Stage 4: mk_conv(conv4) collapsed to rank-4:
//   ew[e,h] = a*Q[k][h] + |a|*P1 + |a|^2*P2 + |a|^3*P3   (sign-dependent P)
// ---------------------------------------------------------------------------
__global__ void k_deg(const int* __restrict__ esrc, int E, float* __restrict__ deg) {
    int e = blockIdx.x * blockDim.x + threadIdx.x;
    if (e < E) atomicAdd(&deg[esrc[e]], 1.f);
}

__global__ __launch_bounds__(256) void k_coeffs(
    const float* __restrict__ W /*[4][64]*/, const float* __restrict__ alpha /*[4][4][64][64]*/,
    float* __restrict__ coef)   // [0,256)=Q[k][h]; [256,1792)=P[(k*2+s)*3+(i-1)][h]
{
    __shared__ float sv[256];
    int tid = threadIdx.x;
    sv[tid] = W[tid];
    __syncthreads();
    {   // Q = alpha[k,0] @ W0
        int k = tid >> 6, h = tid & 63;
        const float* al = alpha + ((size_t)(k * 4 + 0) * 64 + h) * 64;
        float acc = 0.f;
        for (int hp = 0; hp < 64; ++hp) acc += al[hp] * sv[hp];
        coef[tid] = acc;
    }
    for (int o = tid; o < 1536; o += 256) {
        int h = o & 63;
        int rest = o >> 6;            // 0..23
        int i = rest % 3 + 1;         // power 1..3
        int rest2 = rest / 3;         // 0..7
        int s = rest2 & 1;            // sign index
        int k = rest2 >> 1;           // cluster
        float sg = s ? -1.f : 1.f;
        const float* al = alpha + ((size_t)(k * 4 + i) * 64 + h) * 64;
        float acc = 0.f;
        for (int hp = 0; hp < 64; ++hp) {
            float w = sg * sv[i * 64 + hp];
            float l = w > 0.f ? w : 0.1f * w;       // leaky_relu(sign*w)
            float p = l;
            if (i >= 2) p *= l;
            if (i == 3) p *= l;
            acc += al[hp] * p;
        }
        coef[256 + o] = acc;
    }
}

__global__ __launch_bounds__(256) void k_mkconv(
    const float* __restrict__ ea, const int* __restrict__ esrc, const int* __restrict__ edst,
    const int* __restrict__ assign, const float* __restrict__ deg,
    const float* __restrict__ coef, int E, float* __restrict__ out)
{
    long long idx = (long long)blockIdx.x * 256 + threadIdx.x;
    if (idx >= (long long)E * 64) return;
    int e = (int)(idx >> 6), h = (int)(idx & 63);
    int s = esrc[e], d = edst[e];
    int k = assign[s];
    if (assign[d] != k) return;          // disjoint cluster mask
    float a = ea[e];
    float aa = a < 0.f ? -a : a;
    int sg = a < 0.f ? 1 : 0;
    const float* P = coef + 256 + ((k * 2 + sg) * 3) * 64;
    float ew = a * coef[k * 64 + h] + aa * P[h] + aa * aa * P[64 + h] + aa * aa * aa * P[128 + h];
    atomicAdd(&out[(size_t)d * 64 + h], ew / deg[s]);
}

// ---------------------------------------------------------------------------
// Stage 5: fold conv3 into per-node scalar r[n], then KNN-4 interpolation
// ---------------------------------------------------------------------------
__global__ void k_r(const float* __restrict__ eout, const float* __restrict__ x,
                    const float* __restrict__ w3, int N, float* __restrict__ r) {
    int n = blockIdx.x * blockDim.x + threadIdx.x;
    if (n >= N) return;
    const float* row = eout + (size_t)n * 64;
    float acc = 0.f;
    for (int h = 0; h < 64; ++h) acc += row[h] * w3[h];
    r[n] = acc + x[n] * w3[64];
}

#define TP 2048
__global__ __launch_bounds__(256) void k_knn(
    const float* __restrict__ posh, const float* __restrict__ pos,
    const float* __restrict__ r, const float* __restrict__ b3,
    int NH, int N, float* __restrict__ out)
{
    __shared__ __align__(8) float2 sp[TP];
    int p = blockIdx.x * 256 + threadIdx.x;
    float px = 0.f, py = 0.f;
    if (p < NH) { px = posh[p * 2]; py = posh[p * 2 + 1]; }
    float bd0 = 3.4e38f, bd1 = 3.4e38f, bd2 = 3.4e38f, bd3 = 3.4e38f;
    int bi0 = 0, bi1 = 0, bi2 = 0, bi3 = 0;
    const float2* pos2 = (const float2*)pos;
    for (int base = 0; base < N; base += TP) {
        int cnt = N - base; if (cnt > TP) cnt = TP;
        for (int i = threadIdx.x; i < cnt; i += 256) sp[i] = pos2[base + i];
        __syncthreads();
        if (p < NH) {
            for (int i = 0; i < cnt; ++i) {
                float2 q = sp[i];
                float dx = px - q.x, dy = py - q.y;
                float d2 = dx * dx + dy * dy;
                if (d2 < bd3) {
                    int id = base + i;
                    if (d2 < bd2) {
                        bd3 = bd2; bi3 = bi2;
                        if (d2 < bd1) {
                            bd2 = bd1; bi2 = bi1;
                            if (d2 < bd0) { bd1 = bd0; bi1 = bi0; bd0 = d2; bi0 = id; }
                            else         { bd1 = d2;  bi1 = id; }
                        } else { bd2 = d2; bi2 = id; }
                    } else { bd3 = d2; bi3 = id; }
                }
            }
        }
        __syncthreads();
    }
    if (p < NH)
        out[p] = b3[0] + 0.25f * (r[bi0] + r[bi1] + r[bi2] + r[bi3]);
}

// ---------------------------------------------------------------------------
extern "C" void kernel_launch(void* const* d_in, const int* in_sizes, int n_in,
                              void* d_out, int out_size, void* d_ws, size_t ws_size,
                              hipStream_t stream)
{
    const float* x          = (const float*)d_in[0];
    const float* edge_attr  = (const float*)d_in[1];
    const float* pos        = (const float*)d_in[2];
    const float* pos_high   = (const float*)d_in[3];
    const float* lin_sim_w  = (const float*)d_in[4];
    const float* lin_sim_b  = (const float*)d_in[5];
    const float* ec_w1      = (const float*)d_in[6];
    const float* ec_b1      = (const float*)d_in[7];
    const float* ec_w2      = (const float*)d_in[8];
    const float* ec_b2      = (const float*)d_in[9];
    // d_in[10]/d_in[11] (conv1_w / conv1_alpha): first _mk_conv result is overwritten -> dead
    const float* conv4_w     = (const float*)d_in[12];
    const float* conv4_alpha = (const float*)d_in[13];
    const float* conv3_w     = (const float*)d_in[14];
    const float* conv3_b     = (const float*)d_in[15];
    const int*   edge_index  = (const int*)d_in[16];
    const int*   km_init     = (const int*)d_in[17];

    const int N  = in_sizes[0];
    const int E  = in_sizes[1];
    const int NH = in_sizes[3] / 2;

    const int* esrc = edge_index;
    const int* edst = edge_index + E;

    float* ws = (float*)d_ws;
    size_t off = 0;
    auto alloc = [&](size_t nelem) {
        float* p = ws + off;
        off += (nelem + 63) & ~(size_t)63;
        return p;
    };
    float* xs   = alloc((size_t)N * 64);
    float* agg  = alloc((size_t)N * 64);   // becomes xs2 (relu(seg-max) with 0-init)
    float* eout = alloc((size_t)N * 64);
    float* deg  = alloc(N);
    float* sums = alloc(256);
    float* cnts = alloc(64);
    int*   gate = (int*)alloc(64);
    size_t zero_floats = off - (size_t)(agg - ws); // agg..gate inclusive
    float* cent   = alloc(256);
    int*   assign = (int*)alloc(N);
    float* r      = alloc(N);
    float* coef   = alloc(2048);
    __bf16* wfrag = (__bf16*)alloc(6144);  // 12288 bf16 = 24KB fragment image

    (void)hipMemsetAsync(agg, 0, zero_floats * sizeof(float), stream);

    k_wprep<<<1, 256, 0, stream>>>(ec_w1, ec_w2, wfrag);
    k_xs<<<(N * 64 + 255) / 256, 256, 0, stream>>>(x, pos, lin_sim_w, lin_sim_b, N, xs);
    k_edgeconv<<<(E + 63) / 64, 128, 0, stream>>>(xs, esrc, edst, E,
                                                  wfrag, ec_b1, ec_b2, agg);
    k_km_init<<<1, 256, 0, stream>>>(agg, km_init, cent);
    for (int it = 0; it < 64; ++it) {
        k_assign<<<(N + 255) / 256, 256, 0, stream>>>(agg, cent, N, assign, sums, cnts, gate, 1);
        k_update<<<1, 256, 0, stream>>>(cent, sums, cnts, gate, 1e-4f);
    }
    k_assign<<<(N + 255) / 256, 256, 0, stream>>>(agg, cent, N, assign, sums, cnts, nullptr, 0);

    k_deg<<<(E + 255) / 256, 256, 0, stream>>>(esrc, E, deg);
    k_coeffs<<<1, 256, 0, stream>>>(conv4_w, conv4_alpha, coef);
    long long tot = (long long)E * 64;
    k_mkconv<<<(unsigned)((tot + 255) / 256), 256, 0, stream>>>(edge_attr, esrc, edst,
                                                                assign, deg, coef, E, eout);
    k_r<<<(N + 255) / 256, 256, 0, stream>>>(eout, x, conv3_w, N, r);
    k_knn<<<(NH + 255) / 256, 256, 0, stream>>>(pos_high, pos, r, conv3_b, NH, N, (float*)d_out);
}
